// Conv_78022375899305
// MI455X (gfx1250) — matrix operand order repaired and checked by reading, not compiled, not run on hardware
//
#include <hip/hip_runtime.h>
#include <math.h>

// ---------------- problem constants ----------------
#define C_CH   4
#define D_DIR  2
#define S_N    10
#define CD_N   64
#define POS_N  16
#define IN_N   256
#define B_N    64
#define L_N    16
#define N1_N   (L_N * S_N)        // 160
#define N2_N   (L_N * S_N * S_N)  // 1600

// routed-buffer row regions (rows of 256 floats, col = c*64+cd)
#define R0_ROWS (D_DIR * B_N * L_N)    // 2048   ent0 (d==0 -> h)
#define R1_ROWS (D_DIR * B_N * N1_N)   // 20480  ent1
#define R2_ROWS (D_DIR * B_N * N2_N)   // 204800 ent2
#define RI_ROWS (B_N * L_N)            // 1024   item_emb
#define OFF0 0
#define OFF1 (R0_ROWS)                          // 2048
#define OFF2 (OFF1 + R1_ROWS)                   // 22528
#define OFFI (OFF2 + R2_ROWS)                   // 227328
#define TOTAL_ROWS (OFFI + RI_ROWS)             // 228352

#define TILES_PER_BLK 16   // route kernel M-tiles per block (14272 = 892*16)
#define GRP_PER_BLK   8    // agg kernel groups per block (2048,20480 divisible)

typedef __attribute__((ext_vector_type(16))) __bf16 bf16x16;
typedef __attribute__((ext_vector_type(8)))  __bf16 bf16x8;
typedef __attribute__((ext_vector_type(8)))  float  f32x8;

// ---- WMMA fragment loaders (contiguous K order, same for A and B, wave32) ----
// A 16x32 bf16: lane m = lane%16, grp = lane/16.
//   elements 0..15 = row[kb .. kb+15],  kb = kk*32 + grp*16
//   (same contiguous K order as the B fragments)
__device__ __forceinline__ bf16x16 load_a_frag(const __bf16* rowPtr, int kk, int grp) {
    int kb = kk * 32 + grp * 16;
    bf16x8 c0 = *(const bf16x8*)(rowPtr + kb);
    bf16x8 c1 = *(const bf16x8*)(rowPtr + kb + 8);
    bf16x16 a;
#pragma unroll
    for (int e = 0; e < 8; ++e) { a[e] = c0[e]; a[e + 8] = c1[e]; }
    return a;
}
// B 32x16 bf16 stored column-major in LDS (output column's K contiguous):
// lane n = lane%16, grp = lane/16; elements 0..15 = col[kb..kb+15], kb=kk*32+grp*16
__device__ __forceinline__ bf16x16 load_b_frag(const __bf16* colPtr, int kk, int grp) {
    int kb = kk * 32 + grp * 16;
    bf16x8 c0 = *(const bf16x8*)(colPtr + kb);
    bf16x8 c1 = *(const bf16x8*)(colPtr + kb + 8);
    bf16x16 b;
#pragma unroll
    for (int e = 0; e < 8; ++e) { b[e] = c0[e]; b[e + 8] = c1[e]; }
    return b;
}

// =====================================================================
// Kernel 1: route + normalize.  512 thr = 16 waves; each wave owns one
// 16-col N-tile and keeps its 8 B-fragments in VGPRs for the whole
// block; block loops over TILES_PER_BLK M-tiles of 16 gathered rows.
// =====================================================================
__global__ __launch_bounds__(512)
void route_kernel(const float* __restrict__ h, const float* __restrict__ seq,
                  const int* __restrict__ nbr0, const int* __restrict__ nbr1,
                  const int* __restrict__ nbr2, const float* __restrict__ emb,
                  const float* __restrict__ Wr, const float* __restrict__ br,
                  float* __restrict__ routed) {
    __shared__ alignas(16) __bf16 sA[16][IN_N];     // 8 KB
    __shared__ float sOut[16][IN_N];                // 16 KB
    __shared__ float sScale[16][C_CH];
    __shared__ const float* sSrc[16];

    const int t = threadIdx.x;
    const int wave = t >> 5, lane = t & 31, grp = lane >> 4, nl = lane & 15;
    const int n = wave * 16 + nl;         // output column 0..255
    const int c = n >> 6, cd = n & 63;

    // one-time: this wave's B fragments (loop-invariant) into registers
    bf16x16 Bfrag[8];
    {
        const float* bcol = Wr + (size_t)c * IN_N * CD_N + cd;  // W_route[c][k][cd]
#pragma unroll
        for (int kk = 0; kk < 8; ++kk) {
            int k0 = kk * 32 + grp * 16;
#pragma unroll
            for (int e = 0; e < 16; ++e)
                Bfrag[kk][e] = (__bf16)bcol[(size_t)(k0 + e) * CD_N];
        }
    }
    const float bias = br[c * CD_N + cd];

    for (int it = 0; it < TILES_PER_BLK; ++it) {
        const int tile = blockIdx.x * TILES_PER_BLK + it;
        if (t < 16) {  // gather-source pointers for the 16 rows
            long r = (long)tile * 16 + t;
            const float* src;
            if (r < OFF1) {                       // ent0: d==0 -> h, else emb[nbr0]
                int idx = (int)r;
                int d = idx / (B_N * L_N);
                int rem = idx % (B_N * L_N);
                src = (d == 0) ? (h + (size_t)rem * IN_N)
                               : (emb + (size_t)nbr0[idx] * IN_N);
            } else if (r < OFF2) {
                src = emb + (size_t)nbr1[r - OFF1] * IN_N;
            } else if (r < OFFI) {
                src = emb + (size_t)nbr2[r - OFF2] * IN_N;
            } else {
                src = seq + (size_t)(r - OFFI) * IN_N;
            }
            sSrc[t] = src;
        }
        __syncthreads();
        {   // stage A tile: wave w loads row w, lane loads 8 floats -> bf16x8
            const float* src = sSrc[wave] + lane * 8;
            float4 v0 = *(const float4*)(src);
            float4 v1 = *(const float4*)(src + 4);
            bf16x8 p;
            p[0] = (__bf16)v0.x; p[1] = (__bf16)v0.y; p[2] = (__bf16)v0.z; p[3] = (__bf16)v0.w;
            p[4] = (__bf16)v1.x; p[5] = (__bf16)v1.y; p[6] = (__bf16)v1.z; p[7] = (__bf16)v1.w;
            *(bf16x8*)(&sA[wave][lane * 8]) = p;
        }
        __syncthreads();

        f32x8 acc = {};
#pragma unroll
        for (int kk = 0; kk < 8; ++kk) {
            bf16x16 a = load_a_frag(&sA[nl][0], kk, grp);
            acc = __builtin_amdgcn_wmma_f32_16x16x32_bf16(false, a, false, Bfrag[kk],
                                                          (short)0, acc, false, false);
        }
#pragma unroll
        for (int r8 = 0; r8 < 8; ++r8) sOut[r8 + grp * 8][n] = acc[r8] + bias;
        __syncthreads();

        if (t < 64) {  // per (row m, channel cc) L2 norm
            int m = t >> 2, cc = t & 3;
            float ss = 0.f;
#pragma unroll
            for (int k2 = 0; k2 < 64; ++k2) { float v = sOut[m][cc * 64 + k2]; ss += v * v; }
            sScale[m][cc] = 1.0f / fmaxf(sqrtf(ss), 1e-12f);
        }
        __syncthreads();
        for (int i = t; i < 16 * IN_N; i += 512) {
            int m = i >> 8, k = i & 255;
            size_t r = (size_t)tile * 16 + m;
            routed[r * IN_N + k] = sOut[m][k] * sScale[m][k >> 6];
        }
        __syncthreads();  // protect sSrc/sA/sOut reuse next iteration
    }
}

// =====================================================================
// Kernel 2: session context  sess[c][b][cd] = sum_l item[c,b,l,cd]/sum(mask)
// =====================================================================
__global__ __launch_bounds__(64)
void sess_kernel(const float* __restrict__ itemRows,  // routed + OFFI*256
                 const float* __restrict__ mask, float* __restrict__ sess) {
    int blk = blockIdx.x;             // c*B + b
    int c = blk / B_N, b = blk % B_N;
    int k = threadIdx.x;
    float denom = 0.f;
    for (int l = 0; l < L_N; ++l) denom += mask[b * L_N + l];
    float sum = 0.f;
    for (int l = 0; l < L_N; ++l)
        sum += itemRows[((size_t)(b * L_N + l)) * IN_N + c * 64 + k];
    sess[((size_t)c * B_N + b) * 64 + k] = sum / denom;
}

// =====================================================================
// Kernel 3: attention aggregation.  One block = GRP_PER_BLK (d,b,n)
// groups (all 4 channels each); weights staged in LDS once per block.
//   stage1: feats[40(+8)][96] @ w1T -> leakyrelu -> .w2 -> softmax(S)
//   stage2: [self|nv][4(+12)][128] @ w3T -> relu
// =====================================================================
__global__ __launch_bounds__(256)
void agg_kernel(const float* __restrict__ selfRows, const float* __restrict__ neighRows,
                const float* __restrict__ wArr, const float* __restrict__ posArr,
                const float* __restrict__ sess, const float* __restrict__ w1,
                const float* __restrict__ w2, const float* __restrict__ w3,
                float* __restrict__ outRows, int n_per_db) {
    __shared__ alignas(16) __bf16 sF[48][96];    // feats, K padded 81->96
    __shared__ alignas(16) __bf16 sW1[64][96];   // agg_w1^T (n-major)
    __shared__ alignas(16) __bf16 sW3[64][128];  // agg_w3^T (n-major)
    __shared__ alignas(16) __bf16 sF2[16][128];  // stage2 A ([self|nv] rows=c)
    __shared__ float sNeigh[S_N][IN_N];
    __shared__ float sSelf[IN_N];
    __shared__ float sSess[IN_N];
    __shared__ float sW2v[64];
    __shared__ float sS1[48][64];
    __shared__ float sLogit[48];
    __shared__ float sAw[40];
    __shared__ float sNV[IN_N];

    const int t = threadIdx.x;
    const int wave = t >> 5, lane = t & 31, grp = lane >> 4, nl = lane & 15;

    // weights once per block (first __syncthreads below publishes them)
    if (t < 64) sW2v[t] = w2[t];
    for (int i = t; i < 64 * 96; i += 256) {
        int n = i / 96, k = i % 96;
        sW1[n][k] = (__bf16)((k < 81) ? w1[k * 64 + n] : 0.f);
    }
    for (int i = t; i < 64 * 128; i += 256) {
        int n = i >> 7, k = i & 127;
        sW3[n][k] = (__bf16)w3[k * 64 + n];
    }

    for (int gi = 0; gi < GRP_PER_BLK; ++gi) {
        const int g = blockIdx.x * GRP_PER_BLK + gi;
        const int b = (g / n_per_db) % B_N;
        __syncthreads();  // prev iter fully done; weights published on gi==0

        for (int i = t; i < S_N * IN_N; i += 256) {
            int s = i >> 8, k = i & 255;
            sNeigh[s][k] = neighRows[((size_t)g * S_N + s) * IN_N + k];
        }
        sSelf[t] = selfRows[(size_t)g * IN_N + t];
        sSess[t] = sess[((size_t)(t >> 6) * B_N + b) * 64 + (t & 63)];
        __syncthreads();

        // feats rows r = c*10+s : [ sess_c * neigh_s (64) | w (1) | pos (16) | pad ]
        for (int i = t; i < 48 * 96; i += 256) {
            int r = i / 96, k = i % 96;
            float v = 0.f;
            if (r < 40) {
                int c = r / 10, s = r % 10;
                if (k < 64)       v = sSess[c * 64 + k] * sNeigh[s][c * 64 + k];
                else if (k == 64) v = wArr[(size_t)g * S_N + s];
                else if (k < 81)  v = posArr[((size_t)g * S_N + s) * POS_N + (k - 65)];
            }
            sF[r][k] = (__bf16)v;
        }
        __syncthreads();

        // stage 1: 12 tiles (3 M x 4 N), K = 96
        for (int tid = wave; tid < 12; tid += 8) {
            int mt = tid >> 2, nt = tid & 3;
            f32x8 acc = {};
#pragma unroll
            for (int kk = 0; kk < 3; ++kk) {
                bf16x16 a = load_a_frag(&sF[mt * 16 + nl][0], kk, grp);
                bf16x16 bb = load_b_frag(&sW1[nt * 16 + nl][0], kk, grp);
                acc = __builtin_amdgcn_wmma_f32_16x16x32_bf16(false, a, false, bb,
                                                              (short)0, acc, false, false);
            }
            int n = nt * 16 + nl;
            float w2n = sW2v[n];
#pragma unroll
            for (int r8 = 0; r8 < 8; ++r8) {
                int m = mt * 16 + r8 + grp * 8;
                float v = acc[r8];
                v = (v > 0.f) ? v : 0.2f * v;   // leaky_relu(.,0.2)
                sS1[m][n] = v * w2n;
            }
        }
        __syncthreads();
        if (t < 48) {
            float sum = 0.f;
            for (int n = 0; n < 64; ++n) sum += sS1[t][n];
            sLogit[t] = sum;
        }
        __syncthreads();
        if (t < 4) {  // softmax over S within channel t
            float mx = -1e30f;
            for (int s = 0; s < S_N; ++s) mx = fmaxf(mx, sLogit[t * 10 + s]);
            float e[S_N], den = 0.f;
            for (int s = 0; s < S_N; ++s) { e[s] = expf(sLogit[t * 10 + s] - mx); den += e[s]; }
            for (int s = 0; s < S_N; ++s) sAw[t * 10 + s] = e[s] / den;
        }
        __syncthreads();
        {   // nv[col] = sum_s a[c,s] * neigh[s][col]
            int c = t >> 6;
            float nv = 0.f;
            for (int s = 0; s < S_N; ++s) nv += sAw[c * 10 + s] * sNeigh[s][t];
            sNV[t] = nv;
        }
        __syncthreads();
        for (int i = t; i < 16 * 128; i += 256) {
            int r = i >> 7, k = i & 127;
            float v = (r < 4) ? ((k < 64) ? sSelf[r * 64 + k] : sNV[r * 64 + (k - 64)]) : 0.f;
            sF2[r][k] = (__bf16)v;
        }
        __syncthreads();

        // stage 2: 4 N-tiles, M rows 0..3 = channels, K = 128
        if (wave < 4) {
            int nt = wave;
            f32x8 acc = {};
#pragma unroll
            for (int kk = 0; kk < 4; ++kk) {
                bf16x16 a = load_a_frag(&sF2[nl][0], kk, grp);
                bf16x16 bb = load_b_frag(&sW3[nt * 16 + nl][0], kk, grp);
                acc = __builtin_amdgcn_wmma_f32_16x16x32_bf16(false, a, false, bb,
                                                              (short)0, acc, false, false);
            }
#pragma unroll
            for (int r8 = 0; r8 < 8; ++r8) {
                int m = r8 + grp * 8;
                if (m < 4)
                    outRows[(size_t)g * IN_N + m * 64 + nt * 16 + nl] = fmaxf(acc[r8], 0.f);
            }
        }
    }
}

// =====================================================================
// Kernel 4: final gated blend.  16 rows/block, 512 thr = 16 waves.
// =====================================================================
__global__ __launch_bounds__(512)
void final_kernel(const float* __restrict__ entC, const float* __restrict__ hIn,
                  const float* __restrict__ W1, const float* __restrict__ b1,
                  const float* __restrict__ W2, const float* __restrict__ b2,
                  const float* __restrict__ w3v, float* __restrict__ out) {
    __shared__ alignas(16) __bf16 sA1[16][IN_N];
    __shared__ alignas(16) __bf16 sA2[16][IN_N];
    __shared__ float sQ[16][IN_N];
    __shared__ float sAlpha[16];

    const int tile = blockIdx.x;
    const int t = threadIdx.x;

    for (int i = t; i < 16 * IN_N; i += 512) {
        int m = i >> 8, k = i & 255;
        size_t r = (size_t)tile * 16 + m;
        sA1[m][k] = (__bf16)entC[r * IN_N + k];
        sA2[m][k] = (__bf16)hIn[r * IN_N + k];
    }
    __syncthreads();

    const int wave = t >> 5, lane = t & 31, grp = lane >> 4, nl = lane & 15;
    const int n = wave * 16 + nl;
    f32x8 acc = {};
#pragma unroll
    for (int kk = 0; kk < 8; ++kk) {
        bf16x16 a = load_a_frag(&sA1[nl][0], kk, grp);
        bf16x16 bb;
        int k0 = kk * 32 + grp * 16;
#pragma unroll
        for (int e = 0; e < 16; ++e) bb[e] = (__bf16)W1[(size_t)(k0 + e) * IN_N + n];
        acc = __builtin_amdgcn_wmma_f32_16x16x32_bf16(false, a, false, bb,
                                                      (short)0, acc, false, false);
    }
#pragma unroll
    for (int kk = 0; kk < 8; ++kk) {
        bf16x16 a = load_a_frag(&sA2[nl][0], kk, grp);
        bf16x16 bb;
        int k0 = kk * 32 + grp * 16;
#pragma unroll
        for (int e = 0; e < 16; ++e) bb[e] = (__bf16)W2[(size_t)(k0 + e) * IN_N + n];
        acc = __builtin_amdgcn_wmma_f32_16x16x32_bf16(false, a, false, bb,
                                                      (short)0, acc, false, false);
    }
    float bias = b1[n] + b2[n];
#pragma unroll
    for (int r8 = 0; r8 < 8; ++r8) sQ[r8 + grp * 8][n] = acc[r8] + bias;
    __syncthreads();

    if (t < 16) {
        float s = 0.f;
        for (int n2 = 0; n2 < IN_N; ++n2)
            s += w3v[n2] / (1.f + expf(-sQ[t][n2]));
        sAlpha[t] = s;
    }
    __syncthreads();
    for (int i = t; i < 16 * IN_N; i += 512) {
        int m = i >> 8, k = i & 255;
        size_t r = (size_t)tile * 16 + m;
        float al = sAlpha[m];
        out[r * IN_N + k] = al * hIn[r * IN_N + k] + (1.f - al) * entC[r * IN_N + k];
    }
}

// =====================================================================
extern "C" void kernel_launch(void* const* d_in, const int* in_sizes, int n_in,
                              void* d_out, int out_size, void* d_ws, size_t ws_size,
                              hipStream_t stream) {
    (void)in_sizes; (void)n_in; (void)out_size; (void)ws_size;
    const float* h      = (const float*)d_in[0];
    const float* seq    = (const float*)d_in[1];
    const float* mask   = (const float*)d_in[2];
    const int*   nbr0   = (const int*)d_in[3];
    const int*   nbr1   = (const int*)d_in[4];
    const int*   nbr2   = (const int*)d_in[5];
    const float* w_nbr0 = (const float*)d_in[6];
    const float* w_nbr1 = (const float*)d_in[7];
    const float* pos0   = (const float*)d_in[8];
    const float* pos1   = (const float*)d_in[9];
    // d_in[10], d_in[11]: pos_before/pos_after — unused by the reference forward
    const float* emb    = (const float*)d_in[12];
    const float* Wr     = (const float*)d_in[13];
    const float* br     = (const float*)d_in[14];
    const float* aw1    = (const float*)d_in[15];
    const float* aw2    = (const float*)d_in[16];
    const float* aw3    = (const float*)d_in[17];
    const float* l1W    = (const float*)d_in[18];
    const float* l1b    = (const float*)d_in[19];
    const float* l2W    = (const float*)d_in[20];
    const float* l2b    = (const float*)d_in[21];
    const float* l3w    = (const float*)d_in[22];

    // workspace layout (floats): needs ~259 MB
    float* ws     = (float*)d_ws;
    float* routed = ws;                                          // 228352*256
    float* entA   = routed + (size_t)TOTAL_ROWS * IN_N;          // 2048*256
    float* entB   = entA + (size_t)R0_ROWS * IN_N;               // 20480*256
    float* entC   = entB + (size_t)R1_ROWS * IN_N;               // 2048*256
    float* sessB  = entC + (size_t)R0_ROWS * IN_N;               // 4*64*64

    route_kernel<<<TOTAL_ROWS / 16 / TILES_PER_BLK, 512, 0, stream>>>(
        h, seq, nbr0, nbr1, nbr2, emb, Wr, br, routed);
    sess_kernel<<<C_CH * B_N, 64, 0, stream>>>(routed + (size_t)OFFI * IN_N, mask, sessB);
    // hop 0, t=0: self=ent0, neigh=ent1
    agg_kernel<<<R0_ROWS / GRP_PER_BLK, 256, 0, stream>>>(
        routed + (size_t)OFF0 * IN_N, routed + (size_t)OFF1 * IN_N,
        w_nbr0, pos0, sessB, aw1, aw2, aw3, entA, L_N);
    // hop 0, t=1: self=ent1, neigh=ent2
    agg_kernel<<<R1_ROWS / GRP_PER_BLK, 256, 0, stream>>>(
        routed + (size_t)OFF1 * IN_N, routed + (size_t)OFF2 * IN_N,
        w_nbr1, pos1, sessB, aw1, aw2, aw3, entB, N1_N);
    // hop 1, t=0: self=entA, neigh=entB  (nh=1 weights)
    agg_kernel<<<R0_ROWS / GRP_PER_BLK, 256, 0, stream>>>(
        entA, entB, w_nbr0, pos0, sessB,
        aw1 + 81 * 64, aw2 + 64, aw3 + 128 * 64, entC, L_N);
    // final blend over d=0 rows (first B*L rows of entC)
    final_kernel<<<(B_N * L_N) / 16, 512, 0, stream>>>(entC, h, l1W, l1b, l2W, l2b,
                                                       l3w, (float*)d_out);
}
